// EncoderRNN_new_82549271429579
// MI455X (gfx1250) — matrix operand order repaired and checked by reading, not compiled, not run on hardware
//
#include <hip/hip_runtime.h>

typedef __attribute__((ext_vector_type(16))) _Float16 v16h;
typedef __attribute__((ext_vector_type(8)))  float    v8f;

#define H_DIM   128
#define BASKET  200
#define THREADS 256
#define NTILES  24          // 384 rows / 16 rows-per-WMMA-tile, per matrix

__global__ __launch_bounds__(THREADS)
void gru_encoder_step_kernel(const int*   __restrict__ basket,
                             const float* __restrict__ hidden,
                             const float* __restrict__ emb_table,
                             const float* __restrict__ w_ih,
                             const float* __restrict__ w_hh,
                             const float* __restrict__ b_ih,
                             const float* __restrict__ b_hh,
                             float*       __restrict__ out)
{
    __shared__ int      s_basket[BASKET];
    __shared__ float    s_part[2][H_DIM];     // partial gather sums
    __shared__ float    s_g[2 * 3 * H_DIM];   // gi[0..383], gh[384..767]
    __shared__ _Float16 s_vec[2][H_DIM];      // f16 copies: [0]=emb, [1]=h

    const int tid  = threadIdx.x;
    const int lane = tid & 31;
    const int wave = tid >> 5;

    // Prefetch weight matrices into cache while the random gather runs.
    {
        const char* p0 = (const char*)w_ih;
        const char* p1 = (const char*)w_hh;
        const int wbytes = 3 * H_DIM * H_DIM * 4;   // 196608
        for (int off = tid * 128; off < wbytes; off += THREADS * 128) {
            __builtin_prefetch(p0 + off, 0, 0);
            __builtin_prefetch(p1 + off, 0, 0);
        }
    }

    if (tid < BASKET) s_basket[tid] = basket[tid];
    __syncthreads();

    // ---- Phase A: embedding-bag gather-sum over 200 rows ----
    {
        const int col  = tid & (H_DIM - 1);
        const int part = tid >> 7;              // 0 or 1
        float acc = 0.0f;
        #pragma unroll 4
        for (int i = part; i < BASKET; i += 2) {
            const size_t idx = (size_t)s_basket[i];
            acc += emb_table[idx * H_DIM + col];
        }
        s_part[part][col] = acc;
    }
    __syncthreads();
    if (tid < H_DIM) {
        const float e = (s_part[0][tid] + s_part[1][tid]) * (1.0f / (float)H_DIM);
        s_vec[0][tid] = (_Float16)e;            // emb (already / H)
        s_vec[1][tid] = (_Float16)hidden[tid];  // previous hidden
    }
    __syncthreads();

    // ---- Phase B: gi = w_ih@emb, gh = w_hh@h via v_wmma_f32_16x16x32_f16 ----
    // Waves 0-3: w_ih @ emb; waves 4-7: w_hh @ h. 24 tiles per matrix over
    // 4 waves -> 6 tiles/wave; the 4 B-fragments (K=128) are loop-invariant
    // per wave, so hoist them into registers once.
    const int half  = lane >> 4;   // lane group: 0 => lanes 0-15, 1 => lanes 16-31
    const int m     = lane & 15;   // A-matrix row within tile
    const int which = wave >> 2;   // 0: w_ih/emb, 1: w_hh/h
    const int wsub  = wave & 3;    // wave index within its matrix group
    const float* W  = which ? w_hh : w_ih;

    // B fragment: 32x16, vector replicated across all 16 columns.
    // Lanes 0-15 hold K=0..15, lanes 16-31 hold K=16..31 (2 f16 / reg).
    v16h bfrag[4];
    #pragma unroll
    for (int kc = 0; kc < 4; ++kc) {
        #pragma unroll
        for (int j = 0; j < 8; ++j) {
            const int kb = kc * 32 + half * 16 + 2 * j;
            bfrag[kc][2 * j]     = s_vec[which][kb];
            bfrag[kc][2 * j + 1] = s_vec[which][kb + 1];
        }
    }

    for (int t = wsub; t < NTILES; t += 4) {
        const int rowbase = t * 16;
        v8f c = {};
        #pragma unroll
        for (int kc = 0; kc < 4; ++kc) {        // K = 4 x 32 = 128
            // A fragment: 16x32 f16. Lanes 0-15 hold K=0..15, lanes 16-31 hold
            // K=16..31; two packed f16 per 32-bit reg.
            const float* wrow = W + (size_t)(rowbase + m) * H_DIM + kc * 32;
            v16h a;
            #pragma unroll
            for (int j = 0; j < 4; ++j) {
                const int kb = j * 2 + half * 16;
                a[2 * j]     = (_Float16)wrow[kb];
                a[2 * j + 1] = (_Float16)wrow[kb + 1];
            }
            #pragma unroll
            for (int j = 4; j < 8; ++j) {
                const int kb = 8 + (j - 4) * 2 + half * 16;
                a[2 * j]     = (_Float16)wrow[kb];
                a[2 * j + 1] = (_Float16)wrow[kb + 1];
            }
            c = __builtin_amdgcn_wmma_f32_16x16x32_f16(
                    /*neg_a=*/false, a, /*neg_b=*/false, bfrag[kc],
                    /*c_mod=*/(short)0, c, /*reuse_a=*/false, /*reuse_b=*/false);
        }
        // D layout: reg r holds row (r + 8*half), column = lane&15. All columns
        // are identical (replicated B), so let the n==0 lanes (0 and 16) write.
        if ((lane & 15) == 0) {
            #pragma unroll
            for (int r = 0; r < 8; ++r) {
                s_g[which * 384 + rowbase + r + 8 * half] = c[r];
            }
        }
    }
    __syncthreads();

    // ---- Phase C: GRU gate math + write (out, hidden) ----
    if (tid < H_DIM) {
        const int j  = tid;
        const float ir = s_g[j]                 + b_ih[j];
        const float iz = s_g[H_DIM + j]         + b_ih[H_DIM + j];
        const float in = s_g[2 * H_DIM + j]     + b_ih[2 * H_DIM + j];
        const float hr = s_g[384 + j]             + b_hh[j];
        const float hz = s_g[384 + H_DIM + j]     + b_hh[H_DIM + j];
        const float hn = s_g[384 + 2 * H_DIM + j] + b_hh[2 * H_DIM + j];

        const float r = 1.0f / (1.0f + __expf(-(ir + hr)));
        const float z = 1.0f / (1.0f + __expf(-(iz + hz)));
        const float n = tanhf(in + r * hn);
        const float hnew = (1.0f - z) * n + z * hidden[j];

        out[j]         = hnew;   // output [1,1,128]
        out[H_DIM + j] = hnew;   // hidden [1,1,128] (identical for 1-step GRU)
    }
}

extern "C" void kernel_launch(void* const* d_in, const int* in_sizes, int n_in,
                              void* d_out, int out_size, void* d_ws, size_t ws_size,
                              hipStream_t stream) {
    (void)in_sizes; (void)n_in; (void)out_size; (void)d_ws; (void)ws_size;
    const int*   basket    = (const int*)  d_in[0];
    const float* hidden    = (const float*)d_in[1];
    const float* emb_table = (const float*)d_in[2];
    const float* w_ih      = (const float*)d_in[3];
    const float* w_hh      = (const float*)d_in[4];
    const float* b_ih      = (const float*)d_in[5];
    const float* b_hh      = (const float*)d_in[6];
    float*       out       = (float*)d_out;

    gru_encoder_step_kernel<<<1, THREADS, 0, stream>>>(
        basket, hidden, emb_table, w_ih, w_hh, b_ih, b_hh, out);
}
